// BOTNet_2654289789433
// MI455X (gfx1250) — matrix operand order repaired and checked by reading, not compiled, not run on hardware
//
#include <hip/hip_runtime.h>
#include <hip/hip_bf16.h>
#include <math.h>

// ---------------------------------------------------------------------------
// CDNA5 (gfx1250) wave32 WMMA helpers: v_wmma_f32_16x16x32_f16
// LDS is staged in *fragment-native* order so each operand load is two
// ds_load_b128 (32 contiguous bytes per lane) instead of 16x ds_load_u16.
// Mapping is the inverse of the ISA 7.12.2 VGPR layouts:
//   A (16x32, MxK): lane = m + 16*((k>>3)&1), slot = 8*(k>>4) + (k&7)
//   B (32x16, KxN): lane = n + 16*((k>>3)&1), slot = 8*(k>>4) + (k&7)
// ---------------------------------------------------------------------------
typedef __attribute__((ext_vector_type(16))) _Float16 v16h;
typedef __attribute__((ext_vector_type(8)))  float    v8f;

#define WMMA_ACC(c, a, b)                                                     \
  (c) = __builtin_amdgcn_wmma_f32_16x16x32_f16(false, (a), false, (b),        \
                                               (short)0, (c), false, false)

static constexpr float RMAXF   = 5.0f;
static constexpr float PI_F    = 3.14159265358979323846f;
static constexpr float INV_AVG = 1.0f / 16.0f;

// One fragment block = 32 lanes x 16 halves = 512 halves (1KB), 32B/lane.
#define FRAG 512

__device__ inline void fragA_put(_Float16* buf, int m, int k, _Float16 v) {
  int lane = (m & 15) + (((k >> 3) & 1) << 4);
  int slot = ((k >> 4) << 3) + (k & 7);
  buf[lane * 16 + slot] = v;
}
__device__ inline void fragB_put(_Float16* buf, int k, int n, _Float16 v) {
  int lane = (n & 15) + (((k >> 3) & 1) << 4);
  int slot = ((k >> 4) << 3) + (k & 7);
  buf[lane * 16 + slot] = v;
}
__device__ inline v16h frag_ld(const _Float16* buf) {
  const int lane = threadIdx.x & 31;
  return *reinterpret_cast<const v16h*>(buf + lane * 16);
}

// D fragment element r -> (row = r + 8*(lane>=16), col = lane&15)
__device__ inline int d_row(int r) { return r + (((threadIdx.x & 31) >> 4) << 3); }
__device__ inline int d_col()      { return threadIdx.x & 15; }

// Hardware reciprocal (v_rcp_f32, TRANS op — co-executes with VALU) instead
// of the precise-division v_div_scale/v_div_fmas sequence.
__device__ inline float frcp(float x) { return __builtin_amdgcn_rcpf(x); }
__device__ inline float sigf(float x) { return frcp(1.0f + __expf(-x)); }

// ---------------------------------------------------------------------------
// K1: edge geometry: d, u = vec/d, rb (8 bessel*envelope values)
// ---------------------------------------------------------------------------
__global__ __launch_bounds__(256) void k_geom(
    const float* __restrict__ pos, const float* __restrict__ shifts,
    const int* __restrict__ ei, float* __restrict__ d_arr,
    float* __restrict__ u_arr, float* __restrict__ rb_arr, int E_) {
  int e = blockIdx.x * blockDim.x + threadIdx.x;
  if (e >= E_) return;
  int s = ei[e], r = ei[E_ + e];
  float vx = pos[r * 3 + 0] - pos[s * 3 + 0] + shifts[e * 3 + 0];
  float vy = pos[r * 3 + 1] - pos[s * 3 + 1] + shifts[e * 3 + 1];
  float vz = pos[r * 3 + 2] - pos[s * 3 + 2] + shifts[e * 3 + 2];
  float d = sqrtf(vx * vx + vy * vy + vz * vz);
  float inv = frcp(fmaxf(d, 1e-9f));
  d_arr[e] = d;
  u_arr[e * 3 + 0] = vx * inv;
  u_arr[e * 3 + 1] = vy * inv;
  u_arr[e * 3 + 2] = vz * inv;
  float dsafe = fmaxf(d, 1e-6f);
  float uu = d * (1.0f / RMAXF);
  float u2 = uu * uu, u4 = u2 * u2, u5 = u4 * uu, u6 = u5 * uu, u7 = u6 * uu;
  float f = (d < RMAXF) ? (1.0f - 21.0f * u5 + 35.0f * u6 - 15.0f * u7) : 0.0f;
  float cn = sqrtf(2.0f / RMAXF);
  float inv_dsafe = frcp(dsafe);
#pragma unroll
  for (int k = 1; k <= 8; ++k) {
    float a = (float)k * PI_F / RMAXF;
    rb_arr[e * 8 + (k - 1)] = cn * __sinf(a * dsafe) * inv_dsafe * f;
  }
}

// ---------------------------------------------------------------------------
// K2: node embedding h0 = attrs @ W_embed ; e0 contributions
// ---------------------------------------------------------------------------
__global__ __launch_bounds__(256) void k_embed(
    const float* __restrict__ attrs, const float* __restrict__ Wemb,
    const float* __restrict__ aeng, const int* __restrict__ batch,
    float* __restrict__ h0, float* __restrict__ contrib, int N_) {
  int idx = blockIdx.x * blockDim.x + threadIdx.x;
  int n = idx >> 6, c = idx & 63;
  if (n >= N_) return;
  float acc = 0.0f;
#pragma unroll
  for (int z = 0; z < 10; ++z) acc += attrs[n * 10 + z] * Wemb[z * 64 + c];
  h0[idx] = acc;
  if (c == 0) {
    float e0 = 0.0f;
#pragma unroll
    for (int z = 0; z < 10; ++z) e0 += attrs[n * 10 + z] * aeng[z];
    atomicAdd(&contrib[batch[n] * 3 + 0], e0);
  }
}

// ---------------------------------------------------------------------------
// K3 (per layer, fused): radial MLP (WMMA) -> w0 ; msg = w0*hs/16 ;
// scatter atomics into agg[N,64]. Block = 32 edges, 8 wave32 waves.
// ---------------------------------------------------------------------------
__global__ __launch_bounds__(256) void k_fwd_edge(
    const float* __restrict__ rb, const int* __restrict__ ei,
    const float* __restrict__ hprev, const float* __restrict__ W1,
    const float* __restrict__ B1, const float* __restrict__ W2,
    float* __restrict__ agg, int E_) {
  __shared__ _Float16 rbsh[2 * FRAG] __attribute__((aligned(32)));  // A GEMM1 [rt]
  __shared__ _Float16 w1sh[4 * FRAG] __attribute__((aligned(32)));  // B GEMM1 [ct]
  __shared__ _Float16 ash [4 * FRAG] __attribute__((aligned(32)));  // A GEMM2 [rt*2+kb]
  __shared__ _Float16 w2sh[8 * FRAG] __attribute__((aligned(32)));  // B GEMM2 [kb*4+ct]
  __shared__ int snd[32], rcv[32];

  const int tid = threadIdx.x;
  const int eBase = blockIdx.x * 32;
  if (eBase + 32 > E_) return;

  for (int i = tid; i < 1024; i += 256) {  // rb tile 32x32, K padded to 32
    int row = i >> 5, k = i & 31;
    _Float16 v = (k < 8) ? (_Float16)rb[(size_t)(eBase + row) * 8 + k]
                         : (_Float16)0.0f;
    fragA_put(rbsh + (row >> 4) * FRAG, row & 15, k, v);
  }
  for (int i = tid; i < 2048; i += 256) {  // W1 32x64, rows 8..31 zero
    int k = i >> 6, c = i & 63;
    _Float16 v = (k < 8) ? (_Float16)W1[k * 64 + c] : (_Float16)0.0f;
    fragB_put(w1sh + (c >> 4) * FRAG, k, c & 15, v);
  }
  for (int i = tid; i < 4096; i += 256) {  // W2sel 64x64 (every 3rd col)
    int k = i >> 6, c = i & 63;
    fragB_put(w2sh + ((k >> 5) * 4 + (c >> 4)) * FRAG, k & 31, c & 15,
              (_Float16)W2[k * 192 + 3 * c]);
  }
  if (tid < 32) {
    snd[tid] = ei[eBase + tid];
    rcv[tid] = ei[E_ + eBase + tid];
  }
  if (tid == 0 && eBase + 64 <= E_)
    __builtin_prefetch(&rb[(size_t)(eBase + 32) * 8], 0, 1);
  __syncthreads();

  const int wid = tid >> 5;
  const int rt = wid >> 2;  // row tile (0..1): 16 edges
  const int ct = wid & 3;   // col tile (0..3): 16 channels

  // GEMM1: z = rb @ W1 (K=8 padded to 32), then silu -> A of GEMM2
  v8f z = {};
  {
    v16h a = frag_ld(rbsh + rt * FRAG);
    v16h b = frag_ld(w1sh + ct * FRAG);
    WMMA_ACC(z, a, b);
  }
#pragma unroll
  for (int r = 0; r < 8; ++r) {
    int row = rt * 16 + d_row(r);
    int col = ct * 16 + d_col();
    float zv = z[r] + B1[col];
    float s = sigf(zv);
    fragA_put(ash + ((row >> 4) * 2 + (col >> 5)) * FRAG, row & 15, col & 31,
              (_Float16)(zv * s));
  }
  __syncthreads();

  // GEMM2: w0 = silu(z) @ W2sel (K=64)
  v8f w0 = {};
#pragma unroll
  for (int kb = 0; kb < 2; ++kb) {
    v16h a = frag_ld(ash + (rt * 2 + kb) * FRAG);
    v16h b = frag_ld(w2sh + (kb * 4 + ct) * FRAG);
    WMMA_ACC(w0, a, b);
  }

  // Epilogue: msg = w0 * h_prev[sender] / 16, scatter into agg[receiver]
#pragma unroll
  for (int r = 0; r < 8; ++r) {
    int row = rt * 16 + d_row(r);
    int col = ct * 16 + d_col();
    float hv = hprev[(size_t)snd[row] * 64 + col];
    atomicAdd(&agg[(size_t)rcv[row] * 64 + col], w0[r] * hv * INV_AVG);
  }
}

// ---------------------------------------------------------------------------
// K4 (per layer): h = agg @ W_mix[l=0] + attrs@W_skip (WMMA), plus readout
// which==1: e1 = h @ w_read1   which==2: e2 = silu(h@w_m1+b)@w_m2
// ---------------------------------------------------------------------------
__global__ __launch_bounds__(256) void k_mix(
    const float* __restrict__ agg, const float* __restrict__ Wmix0,
    const float* __restrict__ attrs, const float* __restrict__ Wskip,
    const int* __restrict__ batch, float* __restrict__ hout,
    float* __restrict__ contrib, int which, const float* __restrict__ wread1,
    const float* __restrict__ wm1, const float* __restrict__ bm1,
    const float* __restrict__ wm2, int N_) {
  __shared__ _Float16 aash[4 * FRAG] __attribute__((aligned(32)));
  __shared__ _Float16 wmsh[8 * FRAG] __attribute__((aligned(32)));
  __shared__ float wsksh[10 * 64];
  __shared__ float attsh[32 * 10];
  __shared__ float hsh[32 * 64];

  const int tid = threadIdx.x;
  const int nBase = blockIdx.x * 32;
  if (nBase + 32 > N_) return;

  for (int i = tid; i < 2048; i += 256) {
    int row = i >> 6, col = i & 63;
    fragA_put(aash + ((row >> 4) * 2 + (col >> 5)) * FRAG, row & 15, col & 31,
              (_Float16)agg[(size_t)nBase * 64 + i]);
  }
  for (int i = tid; i < 4096; i += 256) {
    int k = i >> 6, c = i & 63;
    fragB_put(wmsh + ((k >> 5) * 4 + (c >> 4)) * FRAG, k & 31, c & 15,
              (_Float16)Wmix0[i]);
  }
  for (int i = tid; i < 640; i += 256) wsksh[i] = Wskip[i];
  for (int i = tid; i < 320; i += 256) {
    int n = i / 10, z = i % 10;
    attsh[i] = attrs[(size_t)(nBase + n) * 10 + z];
  }
  __syncthreads();

  const int wid = tid >> 5;
  const int rt = wid >> 2, ct = wid & 3;
  v8f h = {};
#pragma unroll
  for (int kb = 0; kb < 2; ++kb) {
    v16h a = frag_ld(aash + (rt * 2 + kb) * FRAG);
    v16h b = frag_ld(wmsh + (kb * 4 + ct) * FRAG);
    WMMA_ACC(h, a, b);
  }
#pragma unroll
  for (int r = 0; r < 8; ++r) {
    int row = rt * 16 + d_row(r);
    int col = ct * 16 + d_col();
    float v = h[r];
#pragma unroll
    for (int z = 0; z < 10; ++z) v += attsh[row * 10 + z] * wsksh[z * 64 + col];
    hout[(size_t)(nBase + row) * 64 + col] = v;
    hsh[row * 64 + col] = v;
  }
  __syncthreads();

  if (tid < 32) {
    int n = nBase + tid;
    float e = 0.0f;
    if (which == 1) {
      for (int c = 0; c < 64; ++c) e += hsh[tid * 64 + c] * wread1[c];
    } else {
      for (int k = 0; k < 16; ++k) {
        float zz = bm1[k];
        for (int c = 0; c < 64; ++c) zz += hsh[tid * 64 + c] * wm1[c * 16 + k];
        e += zz * sigf(zz) * wm2[k];
      }
    }
    atomicAdd(&contrib[batch[n] * 3 + which], e);
  }
}

// ---------------------------------------------------------------------------
// B1: backward of readout2: g_h2[n,c] = sum_k silu'(z_k) w_m2[k] w_m1[c,k]
// ---------------------------------------------------------------------------
__global__ __launch_bounds__(128) void k_gh2(
    const float* __restrict__ h2, const float* __restrict__ wm1,
    const float* __restrict__ bm1, const float* __restrict__ wm2,
    float* __restrict__ gh2, int N_) {
  int n = blockIdx.x * blockDim.x + threadIdx.x;
  if (n >= N_) return;
  float hrow[64];
#pragma unroll
  for (int c = 0; c < 64; ++c) hrow[c] = h2[(size_t)n * 64 + c];
  float s[16];
#pragma unroll
  for (int k = 0; k < 16; ++k) {
    float z = bm1[k];
#pragma unroll
    for (int c = 0; c < 64; ++c) z += hrow[c] * wm1[c * 16 + k];
    float sg = sigf(z);
    s[k] = wm2[k] * sg * (1.0f + z * (1.0f - sg));  // silu'
  }
  for (int c = 0; c < 64; ++c) {
    float g = 0.0f;
#pragma unroll
    for (int k = 0; k < 16; ++k) g += s[k] * wm1[c * 16 + k];
    gh2[(size_t)n * 64 + c] = g;
  }
}

// ---------------------------------------------------------------------------
// B2 (per layer): g_agg = (g_h [+ w_read1]) @ W_mix[l=0]^T  (WMMA)
// ---------------------------------------------------------------------------
__global__ __launch_bounds__(256) void k_gagg(
    const float* __restrict__ gh, const float* __restrict__ wread,
    const float* __restrict__ Wmix0, float* __restrict__ gagg, int N_) {
  __shared__ _Float16 gsh [4 * FRAG] __attribute__((aligned(32)));
  __shared__ _Float16 wtsh[8 * FRAG] __attribute__((aligned(32)));
  const int tid = threadIdx.x;
  const int nBase = blockIdx.x * 32;
  if (nBase + 32 > N_) return;

  for (int i = tid; i < 2048; i += 256) {
    int row = i >> 6, col = i & 63;
    float add = wread ? wread[col] : 0.0f;
    fragA_put(gsh + ((row >> 4) * 2 + (col >> 5)) * FRAG, row & 15, col & 31,
              (_Float16)(gh[(size_t)nBase * 64 + i] + add));
  }
  for (int i = tid; i < 4096; i += 256) {
    int d = i >> 6, c = i & 63;  // B[d][c] = Wmix0[c][d]
    fragB_put(wtsh + ((d >> 5) * 4 + (c >> 4)) * FRAG, d & 31, c & 15,
              (_Float16)Wmix0[c * 64 + d]);
  }
  __syncthreads();

  const int wid = tid >> 5;
  const int rt = wid >> 2, ct = wid & 3;
  v8f g = {};
#pragma unroll
  for (int kb = 0; kb < 2; ++kb) {
    v16h a = frag_ld(gsh + (rt * 2 + kb) * FRAG);
    v16h b = frag_ld(wtsh + (kb * 4 + ct) * FRAG);
    WMMA_ACC(g, a, b);
  }
#pragma unroll
  for (int r = 0; r < 8; ++r) {
    int row = rt * 16 + d_row(r);
    int col = ct * 16 + d_col();
    gagg[(size_t)(nBase + row) * 64 + col] = g[r];
  }
}

// ---------------------------------------------------------------------------
// B3 (per layer, fused): recompute radial MLP, backprop through
// message/radial-MLP/bessel down to g_d[e]; optional scatter g_hs -> ghacc.
// Block = 32 edges, WMMA for all 4 GEMM stages.
// ---------------------------------------------------------------------------
__global__ __launch_bounds__(256) void k_bwd_edge(
    const float* __restrict__ rb, const float* __restrict__ d_arr,
    const int* __restrict__ ei, const float* __restrict__ hprev,
    const float* __restrict__ gagg, const float* __restrict__ W1,
    const float* __restrict__ B1, const float* __restrict__ W2,
    float* __restrict__ ghacc, float* __restrict__ g_d, int E_) {
  __shared__ _Float16 rbsh [2 * FRAG] __attribute__((aligned(32)));
  __shared__ _Float16 w1sh [4 * FRAG] __attribute__((aligned(32)));
  __shared__ _Float16 ash  [4 * FRAG] __attribute__((aligned(32)));  // silu(z)
  __shared__ _Float16 w2sh [8 * FRAG] __attribute__((aligned(32)));  // W2sel
  __shared__ _Float16 w2tsh[8 * FRAG] __attribute__((aligned(32)));  // W2sel^T
  __shared__ _Float16 gwsh [4 * FRAG] __attribute__((aligned(32)));  // g_w0
  __shared__ _Float16 gzsh [4 * FRAG] __attribute__((aligned(32)));  // g_z
  __shared__ _Float16 w1tsh[2 * FRAG] __attribute__((aligned(32)));  // W1^T pad
  __shared__ _Float16 spsh[32 * 64];  // silu'(z), row-major
  __shared__ float grbsh[32 * 16];    // g_rb
  __shared__ int snd[32], rcv[32];

  const int tid = threadIdx.x;
  const int eBase = blockIdx.x * 32;
  if (eBase + 32 > E_) return;

  for (int i = tid; i < 1024; i += 256) {
    int row = i >> 5, k = i & 31;
    _Float16 v = (k < 8) ? (_Float16)rb[(size_t)(eBase + row) * 8 + k]
                         : (_Float16)0.0f;
    fragA_put(rbsh + (row >> 4) * FRAG, row & 15, k, v);
  }
  for (int i = tid; i < 2048; i += 256) {
    int k = i >> 6, c = i & 63;
    _Float16 v = (k < 8) ? (_Float16)W1[k * 64 + c] : (_Float16)0.0f;
    fragB_put(w1sh + (c >> 4) * FRAG, k, c & 15, v);
  }
  for (int i = tid; i < 4096; i += 256) {
    int k = i >> 6, c = i & 63;
    fragB_put(w2sh + ((k >> 5) * 4 + (c >> 4)) * FRAG, k & 31, c & 15,
              (_Float16)W2[k * 192 + 3 * c]);
    // transposed: B[cK][kN] = W2sel[kN][cK]
    fragB_put(w2tsh + ((k >> 5) * 4 + (c >> 4)) * FRAG, k & 31, c & 15,
              (_Float16)W2[c * 192 + 3 * k]);
  }
  for (int i = tid; i < 1024; i += 256) {
    int k64 = i >> 4, kb = i & 15;
    _Float16 v = (kb < 8) ? (_Float16)W1[kb * 64 + k64] : (_Float16)0.0f;
    fragB_put(w1tsh + (k64 >> 5) * FRAG, k64 & 31, kb, v);
  }
  if (tid < 32) {
    snd[tid] = ei[eBase + tid];
    rcv[tid] = ei[E_ + eBase + tid];
  }
  __syncthreads();

  const int wid = tid >> 5;
  const int rt = wid >> 2, ct = wid & 3;

  // GEMM1: z = rb @ W1; store silu(z) (frag order) and silu'(z) (row major)
  v8f z = {};
  {
    v16h a = frag_ld(rbsh + rt * FRAG);
    v16h b = frag_ld(w1sh + ct * FRAG);
    WMMA_ACC(z, a, b);
  }
#pragma unroll
  for (int r = 0; r < 8; ++r) {
    int row = rt * 16 + d_row(r);
    int col = ct * 16 + d_col();
    float zv = z[r] + B1[col];
    float s = sigf(zv);
    fragA_put(ash + ((row >> 4) * 2 + (col >> 5)) * FRAG, row & 15, col & 31,
              (_Float16)(zv * s));
    spsh[row * 64 + col] = (_Float16)(s * (1.0f + zv * (1.0f - s)));
  }
  __syncthreads();

  // GEMM2: w0 = silu(z) @ W2sel
  v8f w0 = {};
#pragma unroll
  for (int kb = 0; kb < 2; ++kb) {
    v16h a = frag_ld(ash + (rt * 2 + kb) * FRAG);
    v16h b = frag_ld(w2sh + (kb * 4 + ct) * FRAG);
    WMMA_ACC(w0, a, b);
  }

  // g_msg = g_agg[recv]/16 ; g_w0 = g_msg*hs ; g_hs = g_msg*w0 (scatter)
#pragma unroll
  for (int r = 0; r < 8; ++r) {
    int row = rt * 16 + d_row(r);
    int col = ct * 16 + d_col();
    float gm = gagg[(size_t)rcv[row] * 64 + col] * INV_AVG;
    float hv = hprev[(size_t)snd[row] * 64 + col];
    fragA_put(gwsh + ((row >> 4) * 2 + (col >> 5)) * FRAG, row & 15, col & 31,
              (_Float16)(gm * hv));
    if (ghacc) atomicAdd(&ghacc[(size_t)snd[row] * 64 + col], gm * w0[r]);
  }
  __syncthreads();

  // GEMM3: g_A = g_w0 @ W2sel^T ; g_z = g_A * silu'(z)
  v8f ga = {};
#pragma unroll
  for (int kb = 0; kb < 2; ++kb) {
    v16h a = frag_ld(gwsh + (rt * 2 + kb) * FRAG);
    v16h b = frag_ld(w2tsh + (kb * 4 + ct) * FRAG);
    WMMA_ACC(ga, a, b);
  }
#pragma unroll
  for (int r = 0; r < 8; ++r) {
    int row = rt * 16 + d_row(r);
    int col = ct * 16 + d_col();
    fragA_put(gzsh + ((row >> 4) * 2 + (col >> 5)) * FRAG, row & 15, col & 31,
              (_Float16)(ga[r] * (float)spsh[row * 64 + col]));
  }
  __syncthreads();

  // GEMM4 (waves with ct==0): g_rb = g_z @ W1^T (N padded to 16)
  if (ct == 0) {
    v8f gr = {};
#pragma unroll
    for (int kb = 0; kb < 2; ++kb) {
      v16h a = frag_ld(gzsh + (rt * 2 + kb) * FRAG);
      v16h b = frag_ld(w1tsh + kb * FRAG);
      WMMA_ACC(gr, a, b);
    }
#pragma unroll
    for (int r = 0; r < 8; ++r) {
      int row = rt * 16 + d_row(r);
      grbsh[row * 16 + d_col()] = gr[r];
    }
  }
  __syncthreads();

  // Per-edge: g_d += sum_k g_rb[k] * d(rb_k)/dd
  if (tid < 32) {
    int e = eBase + tid;
    float dd = d_arr[e];
    float dsafe = fmaxf(dd, 1e-6f);
    float uu = dd * (1.0f / RMAXF);
    float u2 = uu * uu, u4 = u2 * u2, u5 = u4 * uu, u6 = u5 * uu, u7 = u6 * uu;
    float f = 1.0f - 21.0f * u5 + 35.0f * u6 - 15.0f * u7;
    float fp = (-105.0f * u4 + 210.0f * u5 - 105.0f * u6) * (1.0f / RMAXF);
    if (dd >= RMAXF) { f = 0.0f; fp = 0.0f; }
    float cn = sqrtf(2.0f / RMAXF);
    float inv_ds = frcp(dsafe);
    float g = 0.0f;
#pragma unroll
    for (int k = 1; k <= 8; ++k) {
      float a = (float)k * PI_F / RMAXF;
      float sn = __sinf(a * dsafe), cs = __cosf(a * dsafe);
      float Bv = cn * sn * inv_ds;
      float dB = cn * (a * cs * inv_ds - sn * inv_ds * inv_ds);
      g += grbsh[tid * 16 + (k - 1)] * (dB * f + Bv * fp);
    }
    g_d[e] += g;
  }
}

// ---------------------------------------------------------------------------
// B4: forces: grad[recv] += g_d*u, grad[send] -= g_d*u; forces = -grad
// ---------------------------------------------------------------------------
__global__ __launch_bounds__(256) void k_forces(
    const float* __restrict__ g_d, const float* __restrict__ u_arr,
    const int* __restrict__ ei, float* __restrict__ forces, int E_) {
  int e = blockIdx.x * blockDim.x + threadIdx.x;
  if (e >= E_) return;
  int s = ei[e], r = ei[E_ + e];
  float g = g_d[e];
#pragma unroll
  for (int j = 0; j < 3; ++j) {
    float gj = g * u_arr[e * 3 + j];
    atomicAdd(&forces[(size_t)r * 3 + j], -gj);
    atomicAdd(&forces[(size_t)s * 3 + j], gj);
  }
}

__global__ void k_total(float* __restrict__ out) {
  int g = threadIdx.x;
  if (g < 64) {
    const float* c = out + 64;
    out[g] = c[g * 3 + 0] + c[g * 3 + 1] + c[g * 3 + 2];
  }
}

// ---------------------------------------------------------------------------
// Host-side launcher
// ---------------------------------------------------------------------------
extern "C" void kernel_launch(void* const* d_in, const int* in_sizes, int n_in,
                              void* d_out, int out_size, void* d_ws,
                              size_t ws_size, hipStream_t stream) {
  const float* pos    = (const float*)d_in[0];
  const float* attrs  = (const float*)d_in[1];
  const float* shifts = (const float*)d_in[2];
  const float* aeng   = (const float*)d_in[3];
  const float* Wemb   = (const float*)d_in[4];
  const float* rw1    = (const float*)d_in[5];   // (2,8,64)
  const float* rb1    = (const float*)d_in[6];   // (2,64)
  const float* rw2    = (const float*)d_in[7];   // (2,64,192)
  const float* Wmix   = (const float*)d_in[8];   // (2,3,64,64)
  const float* Wskip  = (const float*)d_in[9];   // (2,10,64)
  const float* wread1 = (const float*)d_in[10];
  const float* wm1    = (const float*)d_in[11];  // (64,16)
  const float* bm1    = (const float*)d_in[12];
  const float* wm2    = (const float*)d_in[13];
  const int*   ei     = (const int*)d_in[14];    // (2,E)
  const int*   batch  = (const int*)d_in[15];

  const int N_ = in_sizes[0] / 3;
  const int E_ = in_sizes[14] / 2;

  float* out = (float*)d_out;
  float* contrib = out + 64;
  float* forces = out + 64 + 192;

  // Workspace layout (floats)
  float* ws = (float*)d_ws;
  float* d_arr = ws;                         // E
  float* u_arr = d_arr + (size_t)E_;         // 3E
  float* rb    = u_arr + (size_t)3 * E_;     // 8E
  float* h0    = rb + (size_t)8 * E_;        // N*64
  float* h1    = h0 + (size_t)N_ * 64;       // N*64
  float* h2    = h1 + (size_t)N_ * 64;       // N*64
  float* gh2   = h2 + (size_t)N_ * 64;       // N*64
  float* gagg  = gh2 + (size_t)N_ * 64;      // N*64
  // zero-initialized block (contiguous):
  float* agg1   = gagg + (size_t)N_ * 64;    // N*64
  float* agg2   = agg1 + (size_t)N_ * 64;    // N*64
  float* gh1acc = agg2 + (size_t)N_ * 64;    // N*64
  float* g_d    = gh1acc + (size_t)N_ * 64;  // E
  const size_t zeroFloats = (size_t)3 * N_ * 64 + (size_t)E_;

  hipMemsetAsync(d_out, 0, (size_t)out_size * sizeof(float), stream);
  hipMemsetAsync(agg1, 0, zeroFloats * sizeof(float), stream);

  const int eBlocks = E_ / 32;
  const int nBlocks = N_ / 32;

  // Forward
  k_geom<<<(E_ + 255) / 256, 256, 0, stream>>>(pos, shifts, ei, d_arr, u_arr, rb, E_);
  k_embed<<<(N_ * 64 + 255) / 256, 256, 0, stream>>>(attrs, Wemb, aeng, batch, h0, contrib, N_);

  // Layer 0
  k_fwd_edge<<<eBlocks, 256, 0, stream>>>(rb, ei, h0, rw1 + 0 * 512, rb1 + 0 * 64,
                                          rw2 + 0 * 12288, agg1, E_);
  k_mix<<<nBlocks, 256, 0, stream>>>(agg1, Wmix + 0 * 3 * 4096, attrs,
                                     Wskip + 0 * 640, batch, h1, contrib, 1,
                                     wread1, wm1, bm1, wm2, N_);
  // Layer 1
  k_fwd_edge<<<eBlocks, 256, 0, stream>>>(rb, ei, h1, rw1 + 1 * 512, rb1 + 1 * 64,
                                          rw2 + 1 * 12288, agg2, E_);
  k_mix<<<nBlocks, 256, 0, stream>>>(agg2, Wmix + 1 * 3 * 4096, attrs,
                                     Wskip + 1 * 640, batch, h2, contrib, 2,
                                     wread1, wm1, bm1, wm2, N_);

  // Backward
  k_gh2<<<(N_ + 127) / 128, 128, 0, stream>>>(h2, wm1, bm1, wm2, gh2, N_);
  // layer 1 (second) backward: scatter into gh1acc, accumulate g_d
  k_gagg<<<nBlocks, 256, 0, stream>>>(gh2, nullptr, Wmix + 1 * 3 * 4096, gagg, N_);
  k_bwd_edge<<<eBlocks, 256, 0, stream>>>(rb, d_arr, ei, h1, gagg, rw1 + 1 * 512,
                                          rb1 + 1 * 64, rw2 + 1 * 12288, gh1acc,
                                          g_d, E_);
  // layer 0 backward: g_h1 = w_read1 + gh1acc ; no hs scatter (h0 has no pos dep)
  k_gagg<<<nBlocks, 256, 0, stream>>>(gh1acc, wread1, Wmix + 0 * 3 * 4096, gagg, N_);
  k_bwd_edge<<<eBlocks, 256, 0, stream>>>(rb, d_arr, ei, h0, gagg, rw1 + 0 * 512,
                                          rb1 + 0 * 64, rw2 + 0 * 12288, nullptr,
                                          g_d, E_);

  k_forces<<<(E_ + 255) / 256, 256, 0, stream>>>(g_d, u_arr, ei, forces, E_);
  k_total<<<1, 64, 0, stream>>>(out);
}
